// Mamba3LBlock_80161269613360
// MI455X (gfx1250) — matrix operand order repaired and checked by reading, not compiled, not run on hardware
//
#include <hip/hip_runtime.h>

// ---------------------------------------------------------------------------
// Mamba-3 block for MI455X (gfx1250), wave32 + WMMA + async Global->LDS.
// Pipeline: cvt(bf16) -> GEMM_in (bf16 wmma) -> conv+SiLU -> dt/decay ->
//           selective scan (f32 wmma 16x16x4, 1 wave per (b,head),
//           double-buffered GLOBAL_LOAD_ASYNC_TO_LDS for B_t/C_t) ->
//           GEMM_out (bf16 wmma) fused with residual + RMSNorm.
// ---------------------------------------------------------------------------

#define B_      4
#define L_      2048
#define DMODEL  512
#define DSTATE  128
#define RANK    4
#define PCDIM   16
#define DINNER  1024
#define NHEADS  16
#define DIN     3088      // 2*DINNER + 2*BCDIM + NHEADS
#define CONVDIM 2048
#define BCDIM   512
#define M_      (B_ * L_) // 8192

typedef __attribute__((ext_vector_type(2)))  float        v2f;
typedef __attribute__((ext_vector_type(4)))  float        v4f;
typedef __attribute__((ext_vector_type(8)))  float        v8f;
typedef __attribute__((ext_vector_type(16))) __bf16       v16bf;
typedef __attribute__((ext_vector_type(4)))  unsigned int v4u;

#if defined(__has_builtin)
#if __has_builtin(__builtin_amdgcn_wmma_f32_16x16x4_f32)
#define HAVE_WMMA_F32X4 1
#endif
#endif
#ifndef HAVE_WMMA_F32X4
#define HAVE_WMMA_F32X4 0
#endif

// ---------------------------------------------------------------------------
// Async Global -> LDS copy, 16B per lane (ASYNCcnt-tracked; ISA 15.18.3
// opcode 98, GV addressing). LDS address = low 32 bits of the generic
// pointer (LDS aperture keeps the wave-relative byte offset in addr[31:0]).
// ---------------------------------------------------------------------------
__device__ __forceinline__ void async_copy_b128(const float* g, float* lds) {
  unsigned lo = (unsigned)(unsigned long long)lds;
  asm volatile("global_load_async_to_lds_b128 %0, %1, off"
               :: "v"(lo), "v"(g)
               : "memory");
}

__device__ __forceinline__ void wait_asynccnt_le8() {
  asm volatile("s_wait_asynccnt 8" ::: "memory");
}
__device__ __forceinline__ void wait_asynccnt_0() {
  asm volatile("s_wait_asynccnt 0" ::: "memory");
}

// Load a 16-element bf16 fragment as two 16-byte chunks.
__device__ __forceinline__ v16bf load2x8(const __bf16* p0, const __bf16* p1) {
  v16bf r;
  v4u a = *(const v4u*)p0;
  v4u b = *(const v4u*)p1;
  __builtin_memcpy((char*)&r,      &a, 16);
  __builtin_memcpy((char*)&r + 16, &b, 16);
  return r;
}

// D = A(16x4 f32) * B(4x16 f32) + C, wave32.
// A frag: lane (m,hi) holds {A[m][2hi], A[m][2hi+1]}.
// B frag: lane (p,hi) holds {B[2hi][p], B[2hi+1][p]}.
__device__ __forceinline__ v8f wmma_f32x4(v2f a, v2f b, v8f c) {
#if HAVE_WMMA_F32X4
  return __builtin_amdgcn_wmma_f32_16x16x4_f32(false, a, false, b, (short)0, c,
                                               false, false);
#else
  const int lane = threadIdx.x & 31;
  const int hi   = lane >> 4;
  float ox = __shfl(b.x, (lane & 15) | ((1 - hi) << 4));
  float oy = __shfl(b.y, (lane & 15) | ((1 - hi) << 4));
  float b0, b1, b2, b3;
  if (hi == 0) { b0 = b.x; b1 = b.y; b2 = ox; b3 = oy; }
  else         { b0 = ox;  b1 = oy;  b2 = b.x; b3 = b.y; }
  v8f d = c;
  for (int j = 0; j < 8; ++j) {
    int m   = j + 8 * hi;
    float a0 = __shfl(a.x, m);
    float a1 = __shfl(a.y, m);
    float a2 = __shfl(a.x, m + 16);
    float a3 = __shfl(a.y, m + 16);
    d[j] += a0 * b0 + a1 * b1 + a2 * b2 + a3 * b3;
  }
  return d;
#endif
}

__device__ __forceinline__ float silu_f(float x) {
  return x / (1.f + expf(-x));
}

// ---------------------------------------------------------------------------
// Conversion kernels (fp32 -> bf16, weights transposed to K-contiguous).
// ---------------------------------------------------------------------------
__global__ __launch_bounds__(256) void cvt_hidden_kernel(
    const float* __restrict__ x, __bf16* __restrict__ y) {
  size_t i = (size_t)blockIdx.x * 256 + threadIdx.x;   // M_*512
  y[i] = (__bf16)x[i];
}

__global__ __launch_bounds__(256) void cvt_winT_kernel(
    const float* __restrict__ w, __bf16* __restrict__ wt) {
  int i = blockIdx.x * 256 + threadIdx.x;              // DIN*512
  int n = i >> 9, k = i & 511;
  wt[i] = (__bf16)w[(size_t)k * DIN + n];
}

__global__ __launch_bounds__(256) void cvt_woutT_kernel(
    const float* __restrict__ w, __bf16* __restrict__ wt) {
  int i = blockIdx.x * 256 + threadIdx.x;              // 512*1024
  int n = i >> 10, k = i & 1023;
  wt[i] = (__bf16)w[(size_t)k * DMODEL + n];
}

// ---------------------------------------------------------------------------
// GEMM_in: zxbcdt[8192][3088] = hidden_bf16[8192][512] @ W_in.
// WG = 8 waves; wave computes a 64(M) x 16(N) strip (4 accum tiles).
// ---------------------------------------------------------------------------
__global__ __launch_bounds__(256) void gemm_in_kernel(
    const __bf16* __restrict__ A,     // [M_][512]
    const __bf16* __restrict__ BT,    // [3088][512] (W_in transposed)
    float* __restrict__ C) {          // [M_][3088]
  const int lane = threadIdx.x & 31;
  const int wv   = threadIdx.x >> 5;
  const int hi   = lane >> 4;
  const int lo   = lane & 15;
  const int n0   = blockIdx.x * 16;            // 193 tiles
  const int m0   = blockIdx.y * 512 + wv * 64; // 16 blocks

  v8f acc[4];
  for (int t = 0; t < 4; ++t) acc[t] = {};

  for (int k0 = 0; k0 < 512; k0 += 32) {
    const __bf16* bb = BT + (size_t)(n0 + lo) * 512 + k0 + 8 * hi;
    v16bf bf = load2x8(bb, bb + 16);
    for (int t = 0; t < 4; ++t) {
      const __bf16* ab = A + (size_t)(m0 + 16 * t + lo) * 512 + k0 + 8 * hi;
      v16bf af = load2x8(ab, ab + 16);
      acc[t] = __builtin_amdgcn_wmma_f32_16x16x32_bf16(false, af, false, bf,
                                                       (short)0, acc[t],
                                                       false, false);
    }
  }
  for (int t = 0; t < 4; ++t)
    for (int j = 0; j < 8; ++j)
      C[(size_t)(m0 + 16 * t + j + 8 * hi) * DIN + n0 + lo] = acc[t][j];
}

// ---------------------------------------------------------------------------
// Depthwise causal conv (K=4) + bias + SiLU on xBC columns of zxbcdt.
// ---------------------------------------------------------------------------
__global__ __launch_bounds__(256) void conv_silu_kernel(
    const float* __restrict__ zx, const float* __restrict__ cw,
    const float* __restrict__ cb, float* __restrict__ xbc) {
  size_t i = (size_t)blockIdx.x * 256 + threadIdx.x;   // M_*2048
  int c = (int)(i & 2047);
  int m = (int)(i >> 11);
  int l = m & (L_ - 1);
  float acc = cb[c];
  for (int k = 0; k < 4; ++k) {
    int ll = l - 3 + k;
    if (ll >= 0)
      acc += zx[(size_t)(m - l + ll) * DIN + DINNER + c] * cw[c * 4 + k];
  }
  xbc[i] = silu_f(acc);
}

// ---------------------------------------------------------------------------
// dt = softplus(dt_raw + bias), decay = exp(-exp(A_log)*dt).
// ---------------------------------------------------------------------------
__global__ __launch_bounds__(256) void dt_kernel(
    const float* __restrict__ zx, const float* __restrict__ dt_bias,
    const float* __restrict__ A_log, float* __restrict__ dtA,
    float* __restrict__ decA) {
  int i = blockIdx.x * 256 + threadIdx.x;              // M_*16
  int h = i & 15, m = i >> 4;
  float s = zx[(size_t)m * DIN + 3072 + h] + dt_bias[h]; // 3072 = DINNER+CONVDIM
  float dt = (s > 20.f) ? s : log1pf(expf(s));
  dtA[i]  = dt;
  decA[i] = expf(-expf(A_log[h]) * dt);
}

// ---------------------------------------------------------------------------
// Selective scan: one wave per (batch, head).
// State h[128][16] lives in 8 v8f accumulators (WMMA C/D layout).
// B_t(512f)+C_t(512f) are contiguous 4KB per step -> double-buffered into
// LDS with GLOBAL_LOAD_ASYNC_TO_LDS_B128 (8 instrs/step, issued for step
// l+1 while step l computes; s_wait_asynccnt 8 = current buffer resident).
// Per step:  upd = B_t(128x4) @ X_t(4x16)   -> 8x v_wmma_f32_16x16x4_f32
//            h   = dec*h + dt*upd
//            stage h^T into LDS [p][128+4]
//            y   = C_t^T(4x128) @ h(128x16) -> 32x accumulating wmma
// Epilogue fuses +x*D_skip and *silu(z), writes bf16 gated output.
// ---------------------------------------------------------------------------
__global__ __launch_bounds__(32) void scan_kernel(
    const float* __restrict__ zx,    // [M_][3088]  (z source)
    const float* __restrict__ xbc,   // [M_][2048]  post-conv SiLU
    const float* __restrict__ dtA,   // [M_][16]
    const float* __restrict__ decA,  // [M_][16]
    const float* __restrict__ Dskip, // [16]
    __bf16* __restrict__ yg) {       // [M_][1024]  gated output (bf16)
  __shared__ float hbuf[16 * 132];   // h^T staged: [p][128 + 4 pad]
  __shared__ float bc[2][1024];      // double-buffered {B_t, C_t}

  const int bh   = blockIdx.x;
  const int b    = bh >> 4;
  const int h    = bh & 15;
  const int lane = threadIdx.x & 31;
  const int hi   = lane >> 4;
  const int p    = lane & 15;
  const float dsk = Dskip[h];

  const float* xbase = xbc + (size_t)(b << 11) * CONVDIM;

  v8f st[8];
  for (int t = 0; t < 8; ++t) st[t] = {};

  // preload step 0's B/C block (channels 1024..2047, 4KB)
  {
    const float* g = xbase + DINNER;
    for (int i = 0; i < 8; ++i)
      async_copy_b128(g + i * 128 + lane * 4, &bc[0][i * 128 + lane * 4]);
  }

  for (int l = 0; l < L_; ++l) {
    const int row = (b << 11) + l;
    const float* xrow = xbase + (size_t)l * CONVDIM;

    // kick off step l+1's B/C block, then guarantee step l's is resident
    if (l + 1 < L_) {
      const float* g = xrow + CONVDIM + DINNER;
      float* dst = bc[(l + 1) & 1];
      for (int i = 0; i < 8; ++i)
        async_copy_b128(g + i * 128 + lane * 4, dst + i * 128 + lane * 4);
      __builtin_prefetch(xrow + CONVDIM + h * 64, 0, 1);  // next X_t
      wait_asynccnt_le8();
    } else {
      wait_asynccnt_0();
    }
    const float* bcur = bc[l & 1];

    const float dec = decA[row * 16 + h];
    const float dtv = dtA[row * 16 + h];

    // X_t as B-fragment: X[r][p] at channel h*64 + r*16 + p
    v2f xf;
    xf.x = xrow[h * 64 + (2 * hi) * 16 + p];
    xf.y = xrow[h * 64 + (2 * hi + 1) * 16 + p];

    // state update per 16-row n-tile
    for (int t = 0; t < 8; ++t) {
      // B_t[n][r] at bcur[n*4 + r] -> A-frag pair is contiguous in LDS
      v2f bfr = *(const v2f*)(bcur + (16 * t + p) * 4 + 2 * hi);
      v8f zero = {};
      v8f upd  = wmma_f32x4(bfr, xf, zero);
      v8f s    = st[t];
      for (int j = 0; j < 8; ++j) s[j] = dec * s[j] + dtv * upd[j];
      st[t] = s;
      // stage transposed: hbuf[p][16t + 8hi + j] = h[n][p]
      float* dst = hbuf + p * 132 + 16 * t + 8 * hi;
      v4f lo4 = { s[0], s[1], s[2], s[3] };
      v4f hi4 = { s[4], s[5], s[6], s[7] };
      *(v4f*)dst       = lo4;
      *(v4f*)(dst + 4) = hi4;
    }

    // y[r][p] = sum_n C[n][r] * h[n][p]  (split accumulators break the chain)
    const float* crow = bcur + BCDIM;   // C_t[n][r] at crow[n*4 + r]
    const int r = p & 3;                // rows >=4 replicate (never read)
    v8f y0 = {}, y1 = {};
    for (int c = 0; c < 32; c += 2) {
      int n0 = 4 * c;
      v2f cf0; cf0.x = crow[(n0 + 2 * hi) * 4 + r];
               cf0.y = crow[(n0 + 2 * hi + 1) * 4 + r];
      v2f hf0 = *(const v2f*)(hbuf + p * 132 + n0 + 2 * hi);
      y0 = wmma_f32x4(cf0, hf0, y0);

      int n1 = 4 * (c + 1);
      v2f cf1; cf1.x = crow[(n1 + 2 * hi) * 4 + r];
               cf1.y = crow[(n1 + 2 * hi + 1) * 4 + r];
      v2f hf1 = *(const v2f*)(hbuf + p * 132 + n1 + 2 * hi);
      y1 = wmma_f32x4(cf1, hf1, y1);
    }

    // epilogue: lanes 0..15 hold y rows 0..3 in c[0..3]
    if (lane < 16) {
      for (int j = 0; j < 4; ++j) {
        float yv = y0[j] + y1[j];
        int d    = h * 64 + j * 16 + p;
        float xv = xrow[d];
        float zv = zx[(size_t)row * DIN + d];
        yg[(size_t)row * DINNER + d] =
            (__bf16)((yv + xv * dsk) * silu_f(zv));
      }
    }
  }
}

// ---------------------------------------------------------------------------
// GEMM_out fused with residual + RMSNorm.
// WG = 8 waves computes one full 16(M) x 512(N) output row-block.
// ---------------------------------------------------------------------------
__global__ __launch_bounds__(256) void gemm_out_kernel(
    const __bf16* __restrict__ A,    // ygated [M_][1024]
    const __bf16* __restrict__ BT,   // WoutT  [512][1024]
    const float* __restrict__ hid,   // hidden [M_][512]
    float* __restrict__ out) {       // [M_][512]
  __shared__ float resLds[16 * 520];
  __shared__ float red[16 * 16];
  __shared__ float scl[16];

  const int lane = threadIdx.x & 31;
  const int wv   = threadIdx.x >> 5;
  const int hi   = lane >> 4;
  const int lo   = lane & 15;
  const int m0   = blockIdx.x * 16;

  v8f acc[4];
  for (int t = 0; t < 4; ++t) acc[t] = {};

  for (int k0 = 0; k0 < 1024; k0 += 32) {
    const __bf16* ab = A + (size_t)(m0 + lo) * DINNER + k0 + 8 * hi;
    v16bf af = load2x8(ab, ab + 16);
    for (int t = 0; t < 4; ++t) {
      int n = wv * 64 + 16 * t + lo;
      const __bf16* bb = BT + (size_t)n * DINNER + k0 + 8 * hi;
      v16bf bf = load2x8(bb, bb + 16);
      acc[t] = __builtin_amdgcn_wmma_f32_16x16x32_bf16(false, af, false, bf,
                                                       (short)0, acc[t],
                                                       false, false);
    }
  }

  for (int t = 0; t < 4; ++t)
    for (int j = 0; j < 8; ++j) {
      int r = j + 8 * hi;
      int c = wv * 64 + 16 * t + lo;
      resLds[r * 520 + c] = acc[t][j] + hid[(size_t)(m0 + r) * DMODEL + c];
    }
  __syncthreads();

  int r   = threadIdx.x >> 4;
  int seg = threadIdx.x & 15;
  float s = 0.f;
  for (int i = 0; i < 32; ++i) {
    float v = resLds[r * 520 + seg * 32 + i];
    s += v * v;
  }
  red[r * 16 + seg] = s;
  __syncthreads();
  if (threadIdx.x < 16) {
    float tot = 0.f;
    for (int i = 0; i < 16; ++i) tot += red[threadIdx.x * 16 + i];
    scl[threadIdx.x] = rsqrtf(tot / (float)DMODEL + 1e-5f);
  }
  __syncthreads();
  float sc = scl[r];
  for (int i = 0; i < 32; ++i) {
    int c = seg * 32 + i;
    out[(size_t)(m0 + r) * DMODEL + c] = resLds[r * 520 + c] * sc;
  }
}

// ---------------------------------------------------------------------------
// Launcher
// ---------------------------------------------------------------------------
extern "C" void kernel_launch(void* const* d_in, const int* in_sizes, int n_in,
                              void* d_out, int out_size, void* d_ws,
                              size_t ws_size, hipStream_t stream) {
  (void)in_sizes; (void)n_in; (void)out_size; (void)ws_size;
  const float* hs      = (const float*)d_in[0];
  const float* W_in    = (const float*)d_in[1];
  const float* conv_w  = (const float*)d_in[2];
  const float* conv_b  = (const float*)d_in[3];
  const float* dt_bias = (const float*)d_in[4];
  const float* A_log   = (const float*)d_in[5];
  const float* D_skip  = (const float*)d_in[6];
  const float* W_out   = (const float*)d_in[7];
  float* out = (float*)d_out;

  char*  base = (char*)d_ws;
  size_t off  = 0;
  auto carve = [&](size_t bytes) -> void* {
    void* p = base + off;
    off = (off + bytes + 255) & ~(size_t)255;
    return p;
  };
  float*  zx    = (float*)carve((size_t)M_ * DIN * 4);
  float*  xbc   = (float*)carve((size_t)M_ * CONVDIM * 4);
  float*  dtA   = (float*)carve((size_t)M_ * 16 * 4);
  float*  decA  = (float*)carve((size_t)M_ * 16 * 4);
  __bf16* hbf   = (__bf16*)carve((size_t)M_ * DMODEL * 2);
  __bf16* winT  = (__bf16*)carve((size_t)DIN * DMODEL * 2);
  __bf16* woutT = (__bf16*)carve((size_t)DMODEL * DINNER * 2);
  __bf16* yg    = (__bf16*)carve((size_t)M_ * DINNER * 2);

  cvt_hidden_kernel<<<(M_ * DMODEL) / 256, 256, 0, stream>>>(hs, hbf);
  cvt_winT_kernel  <<<(DIN * DMODEL) / 256, 256, 0, stream>>>(W_in, winT);
  cvt_woutT_kernel <<<(DMODEL * DINNER) / 256, 256, 0, stream>>>(W_out, woutT);

  gemm_in_kernel<<<dim3(DIN / 16, M_ / 512), 256, 0, stream>>>(hbf, winT, zx);

  conv_silu_kernel<<<(M_ * CONVDIM) / 256, 256, 0, stream>>>(zx, conv_w,
                                                             conv_b, xbc);
  dt_kernel<<<(M_ * 16) / 256, 256, 0, stream>>>(zx, dt_bias, A_log, dtA,
                                                 decA);

  scan_kernel<<<B_ * NHEADS, 32, 0, stream>>>(zx, xbc, dtA, decA, D_skip, yg);

  gemm_out_kernel<<<M_ / 16, 256, 0, stream>>>(yg, woutT, hs, out);
}
